// Block_57552561766785
// MI455X (gfx1250) — compile-verified, matrix operand order checked by reading, not run on hardware
//
#include <hip/hip_runtime.h>
#include <hip/hip_bf16.h>
#include <stdint.h>

typedef __bf16 bf16;
typedef __attribute__((ext_vector_type(16))) __bf16 v16bf;
typedef __attribute__((ext_vector_type(8)))  float  v8f;
typedef __attribute__((ext_vector_type(4)))  int    v4i;

#define NB   2
#define NT   2048
#define NH   16
#define ND   128
#define NHID 2048
#define NM   (NB * NT)          // 4096 rows
#define SCALE 0.08838834764831845f   // 1/sqrt(128)

#define AS1 __attribute__((address_space(1)))
#define AS3 __attribute__((address_space(3)))

__device__ __forceinline__ v8f wmma_bf16(v16bf a, v16bf b, v8f c) {
  // D = A(16x32 bf16) * B(32x16 bf16) + C(16x16 f32)
  return __builtin_amdgcn_wmma_f32_16x16x32_bf16(false, a, false, b, (short)0, c,
                                                 false, false);
}

// 16-byte global -> LDS copy via the CDNA5 async-to-LDS path (ASYNCcnt).
__device__ __forceinline__ void async_copy16(const void* gptr, void* lptr) {
#if __has_builtin(__builtin_amdgcn_global_load_async_to_lds_b128)
  __builtin_amdgcn_global_load_async_to_lds_b128((AS1 v4i*)gptr, (AS3 v4i*)lptr,
                                                 0, 0);
#else
  *(uint4*)lptr = *(const uint4*)gptr;
#endif
}

__device__ __forceinline__ void wait_async_all() {
  asm volatile("s_wait_asynccnt 0x0" ::: "memory");
}

union AF { v16bf v; uint32_t u[8]; };

// Load a 16x32 bf16 fragment in the CDNA5 16-bit A-matrix layout.
// row (M or N) = lane&15 ; K window split across lane halves (+8 for lanes 16..31).
// Also used for B fragments: B column j has the same per-lane layout as A row j.
__device__ __forceinline__ v16bf load_frag16(const bf16* base, int ld) {
  const int lane  = threadIdx.x & 31;
  const int row   = lane & 15;
  const int khalf = (lane >> 4) << 3;   // 0 or 8
  AF f;
  const bf16* p = base + row * ld;
#pragma unroll
  for (int v = 0; v < 8; ++v) {
    const int kb = ((v < 4) ? (2 * v) : (2 * v + 8)) + khalf; // K0..7 / K16..23 (+8)
    f.u[v] = *(const uint32_t*)(p + kb);                      // packed bf16 pair
  }
  return f.v;
}

// ---------------------------------------------------------------- converts --
__global__ void cast_bf16_kernel(const float* __restrict__ s,
                                 bf16* __restrict__ d, int n) {
  for (int i = blockIdx.x * blockDim.x + threadIdx.x; i < n;
       i += gridDim.x * blockDim.x)
    d[i] = (bf16)s[i];
}

// W[K][N] f32 -> Wt[N][K] bf16
__global__ void cast_trans_kernel(const float* __restrict__ w,
                                  bf16* __restrict__ wt, int K, int N) {
  for (int i = blockIdx.x * blockDim.x + threadIdx.x; i < K * N;
       i += gridDim.x * blockDim.x) {
    const int k = i / N, n = i - k * N;
    wt[(size_t)n * K + k] = (bf16)w[i];
  }
}

// v f32 [M, H*D] -> vt bf16 [B,H,D,T]
__global__ void v_trans_kernel(const float* __restrict__ v,
                               bf16* __restrict__ vt) {
  for (int i = blockIdx.x * blockDim.x + threadIdx.x; i < NM * NHID;
       i += gridDim.x * blockDim.x) {
    const int m = i >> 11, hd = i & (NHID - 1);
    const int h = hd >> 7, d = hd & (ND - 1);
    const int b = m >> 11, t = m & (NT - 1);
    vt[(((size_t)(b * NH + h) * ND) + d) * NT + t] = (bf16)v[i];
  }
}

// ------------------------------------------------------------------- GEMM ---
// C[M,N] f32 = A[M,K] bf16 (row major) * Bt[N,K] bf16 (B transposed)
// Double-buffered LDS; next tile staged via async-to-LDS while WMMAs run.
__global__ __launch_bounds__(256)
void gemm_bf16_kernel(const bf16* __restrict__ A, const bf16* __restrict__ Bt,
                      float* __restrict__ C, int M, int N, int K) {
  __shared__ __align__(16) bf16 As[2][128][32];
  __shared__ __align__(16) bf16 Bs[2][128][32];
  const int tid = threadIdx.x;
  const int m0 = blockIdx.y * 128, n0 = blockIdx.x * 128;
  const int w = tid >> 5, lane = tid & 31;
  const int wm = (w & 3) * 32;      // 4 waves along M
  const int wn = (w >> 2) * 64;     // 2 waves along N

  auto stage = [&](int buf, int kb) {
#pragma unroll
    for (int it = 0; it < 2; ++it) {
      const int i = tid + it * 256;           // 512 16-byte chunks per tile
      const int r = i >> 2, c = (i & 3) * 8;
      async_copy16(&A[(size_t)(m0 + r) * K + kb + c], &As[buf][r][c]);
      async_copy16(&Bt[(size_t)(n0 + r) * K + kb + c], &Bs[buf][r][c]);
    }
  };

  v8f acc[2][4];
#pragma unroll
  for (int i = 0; i < 2; ++i)
#pragma unroll
    for (int j = 0; j < 4; ++j)
      acc[i][j] = (v8f){0.f, 0.f, 0.f, 0.f, 0.f, 0.f, 0.f, 0.f};

  stage(0, 0);
  for (int kb = 0; kb < K; kb += 32) {
    const int buf = (kb >> 5) & 1;
    wait_async_all();        // this wave's tile writes landed in LDS
    __syncthreads();         // => every wave's writes landed; prior reads done
    if (kb + 32 < K) stage(buf ^ 1, kb + 32);   // prefetch overlaps compute

    const v16bf a0 = load_frag16(&As[buf][wm][0], 32);
    const v16bf a1 = load_frag16(&As[buf][wm + 16][0], 32);
#pragma unroll
    for (int j = 0; j < 4; ++j) {
      const v16bf b = load_frag16(&Bs[buf][wn + 16 * j][0], 32);
      acc[0][j] = wmma_bf16(a0, b, acc[0][j]);
      acc[1][j] = wmma_bf16(a1, b, acc[1][j]);
    }
  }

  const int rofs = (lane >> 4) * 8, col = lane & 15;
#pragma unroll
  for (int i = 0; i < 2; ++i)
#pragma unroll
    for (int j = 0; j < 4; ++j)
#pragma unroll
      for (int r = 0; r < 8; ++r) {
        const int row = m0 + wm + 16 * i + rofs + r;
        const int cc = n0 + wn + 16 * j + col;
        C[(size_t)row * N + cc] = acc[i][j][r];
      }
}

// ----------------------------------------------------- RoPE + head RMSNorm --
// src f32 [M, H*D] -> dst bf16 [B,H,T,D]; one wave per (token, head)
__global__ __launch_bounds__(128)
void rope_rms_kernel(const float* __restrict__ src,
                     const float* __restrict__ rms_w, bf16* __restrict__ dst) {
  const int gw = blockIdx.x * 4 + (threadIdx.x >> 5);
  const int lane = threadIdx.x & 31;
  const int m = gw >> 4, h = gw & (NH - 1);
  const int b = m >> 11, t = m & (NT - 1);
  const float* p = src + (size_t)m * NHID + h * ND;

  float out[4];
#pragma unroll
  for (int i = 0; i < 2; ++i) {
    const int j = lane + 32 * i;                 // rotary pair index 0..63
    const float t1 = p[j], t2 = p[j + 64];
    const float inv = __expf(-(float)j * 0.14391156831212787f); // ln(1e4)/64
    const float ang = (float)t * inv;
    const float c = __cosf(ang), s = __sinf(ang);
    out[2 * i]     = t1 * c - t2 * s;            // dim j
    out[2 * i + 1] = t1 * s + t2 * c;            // dim j+64
  }
  float ss = out[0] * out[0] + out[1] * out[1] + out[2] * out[2] + out[3] * out[3];
#pragma unroll
  for (int mk = 1; mk < 32; mk <<= 1) ss += __shfl_xor(ss, mk);
  const float rn = rsqrtf(ss * (1.f / 128.f) + 1e-5f);

  bf16* q = dst + ((size_t)(b * NH + h) * NT + t) * ND;
#pragma unroll
  for (int i = 0; i < 2; ++i) {
    const int j = lane + 32 * i;
    q[j]      = (bf16)(out[2 * i]     * rn * rms_w[j]);
    q[j + 64] = (bf16)(out[2 * i + 1] * rn * rms_w[j + 64]);
  }
}

// -------------------------------------------------------- flash attention ---
// grid = (T/128, B*H); 256 threads; wave w owns q rows q0=qt*128+16w .. +15
__global__ __launch_bounds__(256)
void flash_attn_kernel(const bf16* __restrict__ qbf, const bf16* __restrict__ kbf,
                       const bf16* __restrict__ vt, const float* __restrict__ gate,
                       const float* __restrict__ o_w, bf16* __restrict__ ybf) {
  __shared__ __align__(16) bf16 Ks[2][32][128];     // [buf][k-local][d]
  __shared__ __align__(16) bf16 Vs[2][128][32];     // [buf][d][k-local]
  __shared__ __align__(16) bf16 Pw[8][16][32];      // per-wave P bounce

  const int bh = blockIdx.y, b = bh >> 4, h = bh & (NH - 1);
  const int qt = blockIdx.x;
  const int tid = threadIdx.x, w = tid >> 5, lane = tid & 31;
  const int q0 = qt * 128 + 16 * w;
  const int rofs = (lane >> 4) * 8, cl = lane & 15;

  auto stageKV = [&](int buf, int kb) {
#pragma unroll
    for (int it = 0; it < 2; ++it) {
      const int i = tid + it * 256;
      { const int r = i >> 4, c = (i & 15) * 8;    // Ks: 32 x 128
        async_copy16(&kbf[((size_t)bh * NT + kb + r) * ND + c], &Ks[buf][r][c]); }
      { const int d = i >> 2, c = (i & 3) * 8;     // Vs: 128 x 32
        async_copy16(&vt[((size_t)bh * ND + d) * NT + kb + c], &Vs[buf][d][c]); }
    }
  };

  // Q fragments for this wave's 16 rows, 4 K-windows of 32 over D=128
  const bf16* Qb = qbf + ((size_t)bh * NT + q0) * ND;
  v16bf aq[4];
#pragma unroll
  for (int dw = 0; dw < 4; ++dw) aq[dw] = load_frag16(Qb + 32 * dw, ND);

  v8f o[8];
  float mrow[8], lrow[8];
#pragma unroll
  for (int nt = 0; nt < 8; ++nt) o[nt] = (v8f){0.f,0.f,0.f,0.f,0.f,0.f,0.f,0.f};
#pragma unroll
  for (int r = 0; r < 8; ++r) { mrow[r] = -1e30f; lrow[r] = 0.f; }

  const int kmax = qt * 128 + 128;
  stageKV(0, 0);
  for (int kb = 0; kb < kmax; kb += 32) {
    const int buf = (kb >> 5) & 1;
    wait_async_all();
    __syncthreads();
    if (kb + 32 < kmax) stageKV(buf ^ 1, kb + 32);

    if (kb <= q0 + 15) {
      // S = Q K^T for two 16-wide column tiles
      v8f s[2];
      s[0] = (v8f){0.f,0.f,0.f,0.f,0.f,0.f,0.f,0.f};
      s[1] = (v8f){0.f,0.f,0.f,0.f,0.f,0.f,0.f,0.f};
#pragma unroll
      for (int n = 0; n < 2; ++n)
#pragma unroll
        for (int dw = 0; dw < 4; ++dw) {
          const v16bf bk = load_frag16(&Ks[buf][16 * n][32 * dw], 128);
          s[n] = wmma_bf16(aq[dw], bk, s[n]);
        }
      // scale + causal mask + row max
      float tmax[8];
#pragma unroll
      for (int r = 0; r < 8; ++r) {
        const int qrow = q0 + rofs + r;
#pragma unroll
        for (int n = 0; n < 2; ++n) {
          const int col = kb + 16 * n + cl;
          float v = s[n][r] * SCALE;
          if (col > qrow) v = -1e30f;
          s[n][r] = v;
        }
        tmax[r] = fmaxf(s[0][r], s[1][r]);
      }
#pragma unroll
      for (int mk = 1; mk < 16; mk <<= 1)
#pragma unroll
        for (int r = 0; r < 8; ++r)
          tmax[r] = fmaxf(tmax[r], __shfl_xor(tmax[r], mk));
      // online softmax update
      float alpha[8], rsum[8];
#pragma unroll
      for (int r = 0; r < 8; ++r) {
        const float mn = fmaxf(mrow[r], tmax[r]);
        alpha[r] = __expf(mrow[r] - mn);
        mrow[r] = mn;
        const float p0 = __expf(s[0][r] - mn), p1 = __expf(s[1][r] - mn);
        s[0][r] = p0; s[1][r] = p1;
        rsum[r] = p0 + p1;
      }
#pragma unroll
      for (int mk = 1; mk < 16; mk <<= 1)
#pragma unroll
        for (int r = 0; r < 8; ++r) rsum[r] += __shfl_xor(rsum[r], mk);
#pragma unroll
      for (int r = 0; r < 8; ++r) lrow[r] = lrow[r] * alpha[r] + rsum[r];
#pragma unroll
      for (int nt = 0; nt < 8; ++nt)
#pragma unroll
        for (int r = 0; r < 8; ++r) o[nt][r] *= alpha[r];
      // C-layout -> A-layout transpose for P through per-wave LDS
#pragma unroll
      for (int n = 0; n < 2; ++n)
#pragma unroll
        for (int r = 0; r < 8; ++r)
          Pw[w][rofs + r][16 * n + cl] = (bf16)s[n][r];
      asm volatile("s_wait_dscnt 0x0" ::: "memory");
      const v16bf ap = load_frag16(&Pw[w][0][0], 32);
#pragma unroll
      for (int nt = 0; nt < 8; ++nt) {
        const v16bf bv = load_frag16(&Vs[buf][16 * nt][0], 32);
        o[nt] = wmma_bf16(ap, bv, o[nt]);
      }
    }
  }

  // epilogue: 1/l, RMS-norm over D, SiLU gate, emit bf16
  float ss[8], rn[8];
#pragma unroll
  for (int r = 0; r < 8; ++r) {
    const float il = 1.f / lrow[r];
    ss[r] = 0.f;
#pragma unroll
    for (int nt = 0; nt < 8; ++nt) { o[nt][r] *= il; ss[r] += o[nt][r] * o[nt][r]; }
  }
#pragma unroll
  for (int mk = 1; mk < 16; mk <<= 1)
#pragma unroll
    for (int r = 0; r < 8; ++r) ss[r] += __shfl_xor(ss[r], mk);
#pragma unroll
  for (int r = 0; r < 8; ++r) rn[r] = rsqrtf(ss[r] * (1.f / 128.f) + 1e-5f);

#pragma unroll
  for (int nt = 0; nt < 8; ++nt) {
    const int d = 16 * nt + cl;
    const float ow = o_w[d];
#pragma unroll
    for (int r = 0; r < 8; ++r) {
      const int t = q0 + rofs + r;
      const size_t m = (size_t)b * NT + t;
      const float g = gate[m * NHID + h * ND + d];
      const float y = o[nt][r] * rn[r] * ow * (g / (1.f + __expf(-g)));
      ybf[m * NHID + h * ND + d] = (bf16)y;
    }
  }
}

// ------------------------------------------------------------------- host ---
extern "C" void kernel_launch(void* const* d_in, const int* in_sizes, int n_in,
                              void* d_out, int out_size, void* d_ws, size_t ws_size,
                              hipStream_t stream) {
  (void)in_sizes; (void)n_in; (void)out_size; (void)ws_size;
  const float* x   = (const float*)d_in[0];
  const float* Wq  = (const float*)d_in[1];
  const float* Wk  = (const float*)d_in[2];
  const float* Wv  = (const float*)d_in[3];
  const float* Wg  = (const float*)d_in[4];
  const float* Wo  = (const float*)d_in[5];
  const float* qrw = (const float*)d_in[6];
  const float* krw = (const float*)d_in[7];
  const float* onw = (const float*)d_in[8];
  float* out = (float*)d_out;

  char* p = (char*)d_ws;
  auto alloc = [&](size_t bytes) -> void* {
    void* r = (void*)p; p += (bytes + 255) & ~(size_t)255; return r;
  };
  const size_t MHID = (size_t)NM * NHID;
  bf16* xbf = (bf16*)alloc(MHID * 2);
  bf16* wqT = (bf16*)alloc((size_t)NHID * NHID * 2);
  bf16* wkT = (bf16*)alloc((size_t)NHID * NHID * 2);
  bf16* wvT = (bf16*)alloc((size_t)NHID * NHID * 2);
  bf16* wgT = (bf16*)alloc((size_t)NHID * NHID * 2);
  bf16* woT = (bf16*)alloc((size_t)NHID * NHID * 2);
  float* qf = (float*)alloc(MHID * 4);
  float* kf = (float*)alloc(MHID * 4);
  float* vf = (float*)alloc(MHID * 4);
  float* gf = (float*)alloc(MHID * 4);
  bf16* qbf = (bf16*)alloc(MHID * 2);   // [B,H,T,D]
  bf16* kbf = (bf16*)alloc(MHID * 2);   // [B,H,T,D]
  bf16* vtb = (bf16*)alloc(MHID * 2);   // [B,H,D,T]
  bf16* ybf = (bf16*)alloc(MHID * 2);   // gated attention out [M, H*D]

  cast_bf16_kernel<<<2048, 256, 0, stream>>>(x, xbf, (int)MHID);
  cast_trans_kernel<<<2048, 256, 0, stream>>>(Wq, wqT, NHID, NHID);
  cast_trans_kernel<<<2048, 256, 0, stream>>>(Wk, wkT, NHID, NHID);
  cast_trans_kernel<<<2048, 256, 0, stream>>>(Wv, wvT, NHID, NHID);
  cast_trans_kernel<<<2048, 256, 0, stream>>>(Wg, wgT, NHID, NHID);
  cast_trans_kernel<<<2048, 256, 0, stream>>>(Wo, woT, NHID, NHID);

  dim3 gg(NHID / 128, NM / 128);
  gemm_bf16_kernel<<<gg, 256, 0, stream>>>(xbf, wqT, qf, NM, NHID, NHID);
  gemm_bf16_kernel<<<gg, 256, 0, stream>>>(xbf, wkT, kf, NM, NHID, NHID);
  gemm_bf16_kernel<<<gg, 256, 0, stream>>>(xbf, wvT, vf, NM, NHID, NHID);
  gemm_bf16_kernel<<<gg, 256, 0, stream>>>(xbf, wgT, gf, NM, NHID, NHID);

  rope_rms_kernel<<<(NM * NH) / 4, 128, 0, stream>>>(qf, qrw, qbf);
  rope_rms_kernel<<<(NM * NH) / 4, 128, 0, stream>>>(kf, krw, kbf);
  v_trans_kernel<<<4096, 256, 0, stream>>>(vf, vtb);

  dim3 fg(NT / 128, NB * NH);
  flash_attn_kernel<<<fg, 256, 0, stream>>>(qbf, kbf, vtb, gf, onw, ybf);

  gemm_bf16_kernel<<<gg, 256, 0, stream>>>(ybf, woT, out, NM, NHID, NHID);
}
